// ArcosLossWithWeights_910533067140
// MI455X (gfx1250) — matrix operand order repaired and checked by reading, not compiled, not run on hardware
//
#include <hip/hip_runtime.h>
#include <hip/hip_bf16.h>

// ArcFace loss on MI455X (gfx1250):
//   - W (307 MB fp32) streamed from HBM exactly once: grid over classes,
//     loop over all embedding rows inside the workgroup (A_f16 = 1.5 MB,
//     L2-resident across the 3125 workgroups).
//   - Matmul via v_wmma_f32_16x16x32_f16 (f16 in, fp32 accumulate).
//   - 2x2 register tiling per wave (32x32 output per wave): each A fragment
//     feeds 2 n-subtiles, each B fragment feeds 2 m-subtiles -> 1 fragment
//     load per WMMA instead of 2.
//   - Softmax uses the bound |logit| <= SCALE=20 -> fixed-shift exp, no
//     max pass; deterministic per-WG partial sums (no float atomics).

typedef __attribute__((ext_vector_type(16))) _Float16 v16h;
typedef __attribute__((ext_vector_type(8)))  _Float16 v8h;
typedef __attribute__((ext_vector_type(8)))  float    v8f;

#define B_ROWS 1024
#define D_DIM  768
#define C_CLS  100000
#define N_BLK  32            // classes per workgroup (100000 = 32 * 3125)
#define N_WGS  (C_CLS / N_BLK)
#define M_CHUNK 256          // rows per iteration: 8 waves * 32 rows
#define LDW    776           // f16 LDS stride (768 + 8), keeps 16B alignment

#define MARGIN 0.4f
#define SCALE  20.0f
#define EPS    1e-7f

// ---------------------------------------------------------------- kernel 1
// Normalize embedding rows, emit f16 row-major A (one block per row).
__global__ void arc_norm_a(const float* __restrict__ emb,
                           _Float16* __restrict__ a16) {
  __shared__ float red[256];
  const int row = blockIdx.x;
  const int t   = threadIdx.x;
  const float* r = emb + (size_t)row * D_DIM;

  float s = 0.f;
  #pragma unroll
  for (int j = 0; j < 3; ++j) {          // 768 = 256 * 3, stride-256 coalesced
    float v = r[t + j * 256];
    s += v * v;
  }
  red[t] = s;
  __syncthreads();
  for (int off = 128; off > 0; off >>= 1) {
    if (t < off) red[t] += red[t + off];
    __syncthreads();
  }
  const float scale = 1.0f / sqrtf(red[0]);
  _Float16* o = a16 + (size_t)row * D_DIM;
  #pragma unroll
  for (int j = 0; j < 3; ++j)
    o[t + j * 256] = (_Float16)(r[t + j * 256] * scale);
}

// ---------------------------------------------------------------- kernel 2
// Per workgroup: normalize 32 W rows into LDS (f16), then WMMA against all
// 1024 embedding rows with 2x2 register tiling per wave; fused
// clamp/margin/exp epilogue; per-WG partial softmax sums + label logits.
__global__ __launch_bounds__(256)
void arc_gemm(const _Float16* __restrict__ a16,
              const float*    __restrict__ W,
              const int*      __restrict__ labels,
              float*          __restrict__ labelLogit,
              float*          __restrict__ partial) {
  __shared__ __align__(16) _Float16 Wl[N_BLK][LDW];
  __shared__ float sq[N_BLK][8];
  __shared__ float wscale[N_BLK];
  __shared__ int   labLds[M_CHUNK];

  const int wg   = blockIdx.x;
  const int n0   = wg * N_BLK;
  const int tid  = threadIdx.x;
  const int wave = tid >> 5;      // 8 waves, each owns 32 rows x 32 cols
  const int lane = tid & 31;
  const int half = lane >> 4;     // 0: lanes 0-15, 1: lanes 16-31
  const int l    = lane & 15;

  // ---- phase 1: row sums of squares for this WG's 32 W rows
  {
    const int r = tid >> 3, p = tid & 7;
    const float* wrow = W + (size_t)(n0 + r) * D_DIM + p * 96;
    float s = 0.f;
    #pragma unroll 8
    for (int i = 0; i < 96; ++i) s += wrow[i] * wrow[i];
    sq[r][p] = s;
  }
  __syncthreads();
  if (tid < N_BLK) {
    float s = 0.f;
    #pragma unroll
    for (int j = 0; j < 8; ++j) s += sq[tid][j];
    wscale[tid] = 1.0f / sqrtf(s);
  }
  __syncthreads();
  // normalize + convert to f16 into LDS (second W read hits L2/WGP$)
  for (int idx = tid; idx < N_BLK * (D_DIM / 4); idx += 256) {
    const int r = idx / (D_DIM / 4), k4 = idx % (D_DIM / 4);
    const float4 v = ((const float4*)(W + (size_t)(n0 + r) * D_DIM))[k4];
    const float sc = wscale[r];
    Wl[r][k4 * 4 + 0] = (_Float16)(v.x * sc);
    Wl[r][k4 * 4 + 1] = (_Float16)(v.y * sc);
    Wl[r][k4 * 4 + 2] = (_Float16)(v.z * sc);
    Wl[r][k4 * 4 + 3] = (_Float16)(v.w * sc);
  }
  __syncthreads();

  // ---- phase 2: sweep all 1024 embedding rows in chunks of 256
  for (int m0 = 0; m0 < B_ROWS; m0 += M_CHUNK) {
    labLds[tid] = labels[m0 + tid];
    __syncthreads();

    const int mbase_w = m0 + wave * 32;        // this wave's 32 rows
    const _Float16* Ab0 = a16 + (size_t)(mbase_w + l) * D_DIM      + half * 8;
    const _Float16* Ab1 = a16 + (size_t)(mbase_w + 16 + l) * D_DIM + half * 8;
    const _Float16* Bb0 = &Wl[l][half * 8];
    const _Float16* Bb1 = &Wl[16 + l][half * 8];

    v8f acc[2][2];
    #pragma unroll
    for (int mi = 0; mi < 2; ++mi)
      #pragma unroll
      for (int ni = 0; ni < 2; ++ni)
        acc[mi][ni] = (v8f){0.f, 0.f, 0.f, 0.f, 0.f, 0.f, 0.f, 0.f};

    #pragma unroll 2
    for (int k0 = 0; k0 < D_DIM; k0 += 32) {
      // A fragments (16x32 f16), documented per-lane K chunks
      const v16h a0 = __builtin_shufflevector(
          *(const v8h*)(Ab0 + k0), *(const v8h*)(Ab0 + k0 + 16),
          0, 1, 2, 3, 4, 5, 6, 7, 8, 9, 10, 11, 12, 13, 14, 15);
      const v16h a1 = __builtin_shufflevector(
          *(const v8h*)(Ab1 + k0), *(const v8h*)(Ab1 + k0 + 16),
          0, 1, 2, 3, 4, 5, 6, 7, 8, 9, 10, 11, 12, 13, 14, 15);
      // B fragments (32x16 f16) from LDS
      const v16h b0 = __builtin_shufflevector(
          *(const v8h*)(Bb0 + k0), *(const v8h*)(Bb0 + k0 + 16),
          0, 1, 2, 3, 4, 5, 6, 7, 8, 9, 10, 11, 12, 13, 14, 15);
      const v16h b1 = __builtin_shufflevector(
          *(const v8h*)(Bb1 + k0), *(const v8h*)(Bb1 + k0 + 16),
          0, 1, 2, 3, 4, 5, 6, 7, 8, 9, 10, 11, 12, 13, 14, 15);
      acc[0][0] = __builtin_amdgcn_wmma_f32_16x16x32_f16(
          false, a0, false, b0, (short)0, acc[0][0], false, false);
      acc[0][1] = __builtin_amdgcn_wmma_f32_16x16x32_f16(
          false, a0, false, b1, (short)0, acc[0][1], false, false);
      acc[1][0] = __builtin_amdgcn_wmma_f32_16x16x32_f16(
          false, a1, false, b0, (short)0, acc[1][0], false, false);
      acc[1][1] = __builtin_amdgcn_wmma_f32_16x16x32_f16(
          false, a1, false, b1, (short)0, acc[1][1], false, false);
    }

    // epilogue: cos -> (margin) -> logit -> exp(logit - SCALE); the wave
    // owns both n-subtiles of its rows, so the 32-column row sum forms
    // entirely in registers + lane shuffles.
    #pragma unroll
    for (int mi = 0; mi < 2; ++mi) {
      #pragma unroll
      for (int r = 0; r < 8; ++r) {
        const int lrow = mi * 16 + half * 8 + r;   // within wave's 32 rows
        const int grow = mbase_w + lrow;
        const int lab  = labLds[wave * 32 + lrow];
        float esum = 0.f;
        #pragma unroll
        for (int ni = 0; ni < 2; ++ni) {
          float c = acc[mi][ni][r];
          c = fminf(fmaxf(c, -1.0f + EPS), 1.0f - EPS);
          float logit = c * SCALE;
          if (lab == n0 + ni * 16 + l) {
            logit = cosf(acosf(c) + MARGIN) * SCALE;
            labelLogit[grow] = logit;              // unique writer per row
          }
          esum += __expf(logit - SCALE);
        }
        esum += __shfl_xor(esum, 1);               // reduce over 16 N-lanes
        esum += __shfl_xor(esum, 2);               // (masks < 16 stay in half)
        esum += __shfl_xor(esum, 4);
        esum += __shfl_xor(esum, 8);
        if (l == 0)
          partial[(size_t)grow * N_WGS + wg] = esum;
      }
    }
    __syncthreads();   // protect labLds before next chunk's refill
  }
}

// ---------------------------------------------------------------- kernel 3
// Reduce 3125 partials per row -> nll[row] = SCALE + log(sum) - labelLogit.
__global__ void arc_rowfinal(const float* __restrict__ partial,
                             const float* __restrict__ labelLogit,
                             float* __restrict__ nll) {
  __shared__ float red[256];
  const int m = blockIdx.x;
  const int t = threadIdx.x;
  const float* p = partial + (size_t)m * N_WGS;
  float s = 0.f;
  for (int wg = t; wg < N_WGS; wg += 256) s += p[wg];
  red[t] = s;
  __syncthreads();
  for (int off = 128; off > 0; off >>= 1) {
    if (t < off) red[t] += red[t + off];
    __syncthreads();
  }
  if (t == 0) nll[m] = SCALE + logf(red[0]) - labelLogit[m];
}

// ---------------------------------------------------------------- kernel 4
__global__ void arc_mean(const float* __restrict__ nll, float* __restrict__ out) {
  __shared__ float red[256];
  const int t = threadIdx.x;
  float s = 0.f;
  #pragma unroll
  for (int j = 0; j < 4; ++j) s += nll[t + j * 256];
  red[t] = s;
  __syncthreads();
  for (int off = 128; off > 0; off >>= 1) {
    if (t < off) red[t] += red[t + off];
    __syncthreads();
  }
  if (t == 0) out[0] = red[0] * (1.0f / (float)B_ROWS);
}

// ---------------------------------------------------------------- launcher
extern "C" void kernel_launch(void* const* d_in, const int* in_sizes, int n_in,
                              void* d_out, int out_size, void* d_ws, size_t ws_size,
                              hipStream_t stream) {
  const float* emb    = (const float*)d_in[0];   // [1024, 768] fp32
  const int*   labels = (const int*)d_in[1];     // [1024] int
  const float* W      = (const float*)d_in[2];   // [100000, 768] fp32
  float* out = (float*)d_out;

  char* ws = (char*)d_ws;
  _Float16* a16        = (_Float16*)(ws);                      // 1,572,864 B
  float*    labelLogit = (float*)(ws + 1572864);               //     4,096 B
  float*    partial    = (float*)(ws + 1576960);               // 12,800,000 B
  float*    nll        = (float*)(ws + 14376960);              //     4,096 B

  arc_norm_a  <<<B_ROWS, 256, 0, stream>>>(emb, a16);
  arc_gemm    <<<N_WGS,  256, 0, stream>>>(a16, W, labels, labelLogit, partial);
  arc_rowfinal<<<B_ROWS, 256, 0, stream>>>(partial, labelLogit, nll);
  arc_mean    <<<1, 256, 0, stream>>>(nll, out);
}
